// DLASSO_GNNHyp3_10677288698537
// MI455X (gfx1250) — matrix operand order repaired and checked by reading, not compiled, not run on hardware
//
#include <hip/hip_runtime.h>
#include <math.h>

// ---------------- problem constants (match reference) ----------------
#define PP   50
#define MM   512
#define NN   1024
#define HH   256
#define H4   1024
#define BB   8
#define EE   400
#define KK   5
#define TWO_N 2048
#define BPN  (BB*PP*NN)   // 409600

typedef _Float16 f16;
typedef __attribute__((ext_vector_type(2)))  _Float16 v2h;
typedef __attribute__((ext_vector_type(16))) _Float16 v16h;
typedef __attribute__((ext_vector_type(8)))  float    v8f;
typedef __attribute__((ext_vector_type(4)))  int      v4i;

// gfx1250 async global->LDS staging (ASYNCcnt), feature-probed.
// Prototype (from hipcc diagnostic): (v4i AS1*, v4i AS3*, imm int, imm int).
#if __has_builtin(__builtin_amdgcn_global_load_async_to_lds_b128)
#define USE_ASYNC_LDS 1
typedef __attribute__((address_space(1))) v4i g_v4i;
typedef __attribute__((address_space(3))) v4i l_v4i;
#else
#define USE_ASYNC_LDS 0
#endif

__device__ __forceinline__ void wait_asynccnt0() {
#if __has_builtin(__builtin_amdgcn_s_wait_asynccnt)
    __builtin_amdgcn_s_wait_asynccnt(0);
#else
    asm volatile("s_wait_asynccnt 0x0" ::: "memory");
#endif
}

// ---------------- generic strided WMMA GEMM with split-K -------------
// C[m,n] (+)= sum_k A[m,k] * W[k,n], atomically accumulated (C pre-zeroed).
#define BM 64
#define BN 64
#define BK 32
#define LDT 40   // LDS row pitch in halfs

// Cooperative tile load: each thread converts 4 K-pairs of A and W and
// stores them as packed b32. G=false -> full tile, no guards (fast path);
// G=true  -> branch-free clamped loads (v_cndmask, no exec waterfalls).
template <bool G>
__device__ __forceinline__ void load_tile_pair(
    f16* __restrict__ As, f16* __restrict__ Bs,
    const float* __restrict__ Ap, const float* __restrict__ Wp,
    int tid, int m0, int n0, int kk, int k1, int Mg, int Ng,
    long long a_rs, long long a_cs, long long w_rs, long long w_cs)
{
    #pragma unroll
    for (int i = 0; i < 4; ++i) {
        int q  = tid + i * 256;     // pair index over 64x16 pairs
        int r  = q >> 4;            // row within tile (A: m, W: n)
        int c2 = (q & 15) << 1;     // k offset within tile (even)
        long long ia0 = (long long)(m0 + r) * a_rs + (long long)(kk + c2) * a_cs;
        long long iw0 = (long long)(kk + c2) * w_rs + (long long)(n0 + r) * w_cs;
        float a0, a1, w0, w1;
        if constexpr (!G) {
            a0 = Ap[ia0];
            a1 = Ap[ia0 + a_cs];
            w0 = Wp[iw0];
            w1 = Wp[iw0 + w_rs];
        } else {
            bool am = (m0 + r) < Mg;
            bool ka = (kk + c2) < k1;
            bool kb = (kk + c2 + 1) < k1;
            bool v0 = am && ka, v1 = am && kb;
            a0 = Ap[v0 ? ia0 : 0];          a0 = v0 ? a0 : 0.f;
            a1 = Ap[v1 ? (ia0 + a_cs) : 0]; a1 = v1 ? a1 : 0.f;
            bool wn = (n0 + r) < Ng;
            bool u0 = wn && ka, u1 = wn && kb;
            w0 = Wp[u0 ? iw0 : 0];          w0 = u0 ? w0 : 0.f;
            w1 = Wp[u1 ? (iw0 + w_rs) : 0]; w1 = u1 ? w1 : 0.f;
        }
        v2h pa = { (f16)a0, (f16)a1 };
        v2h pw = { (f16)w0, (f16)w1 };
        *(v2h*)(&As[r * LDT + c2]) = pa;
        *(v2h*)(&Bs[r * LDT + c2]) = pw;
    }
}

__global__ __launch_bounds__(256)
void wmma_gemm(const float* __restrict__ Ag, const float* __restrict__ Wg,
               float* __restrict__ Cg,
               int Mg, int Ng, int Kg,
               long long a_base, long long a_rs, long long a_cs, long long a_bs,
               long long w_base, long long w_rs, long long w_cs, long long w_bs,
               long long c_base, long long c_rs, long long c_cs, long long c_bs,
               int ksplit)
{
    __shared__ f16 As[BM * LDT];
    __shared__ f16 Bs[BN * LDT];

    const int tid  = threadIdx.x;
    const int lane = tid & 31;
    const int wave = tid >> 5;

    const int z   = blockIdx.z;
    const int kc  = z % ksplit;
    const int bat = z / ksplit;
    const int kchunk = (Kg + ksplit - 1) / ksplit;
    const int k0 = kc * kchunk;
    const int k1 = (k0 + kchunk < Kg) ? (k0 + kchunk) : Kg;

    const int m0 = blockIdx.y * BM;
    const int n0 = blockIdx.x * BN;

    const float* Ap = Ag + a_base + (long long)bat * a_bs;
    const float* Wp = Wg + w_base + (long long)bat * w_bs;
    float*       Cp = Cg + c_base + (long long)bat * c_bs;

    // 8 waves -> 4x2 tiling: each wave owns a 16x32 patch (two 16x16 WMMAs)
    const int wm = (wave & 3) * 16;
    const int wn = (wave >> 2) * 32;

    v8f acc0 = {};
    v8f acc1 = {};

    const bool mn_full = (m0 + BM <= Mg) && (n0 + BN <= Ng);

    for (int kk = k0; kk < k1; kk += BK) {
        if (mn_full && (kk + BK <= k1))
            load_tile_pair<false>(As, Bs, Ap, Wp, tid, m0, n0, kk, k1, Mg, Ng,
                                  a_rs, a_cs, w_rs, w_cs);
        else
            load_tile_pair<true>(As, Bs, Ap, Wp, tid, m0, n0, kk, k1, Mg, Ng,
                                 a_rs, a_cs, w_rs, w_cs);
        // speculative prefetch of next A tile (global_prefetch_b8)
        if (kk + BK < k1)
            __builtin_prefetch(Ap + (long long)m0 * a_rs + (long long)(kk + BK) * a_cs, 0, 1);
        __syncthreads();

        // Fragments in contiguous k order (k = 16*(lane>>4) + j), same for A and B.
        v16h af, bf0, bf1;
        const int ml    = wm + (lane & 15);
        const int khalf = (lane >> 4) * 16;     // lanes 16-31 hold K=16..31
        #pragma unroll
        for (int j = 0; j < 16; ++j) {
            int k = khalf + j;
            af[j] = As[ml * LDT + k];
        }
        const int nl0 = wn + (lane & 15);
        const int nl1 = nl0 + 16;
        const int kb  = (lane >> 4) * 16;       // lanes 16-31 hold K=16..31
        #pragma unroll
        for (int j = 0; j < 16; ++j) {
            bf0[j] = Bs[nl0 * LDT + kb + j];
            bf1[j] = Bs[nl1 * LDT + kb + j];
        }

        acc0 = __builtin_amdgcn_wmma_f32_16x16x32_f16(false, af, false, bf0,
                                                      (short)0, acc0, false, false);
        acc1 = __builtin_amdgcn_wmma_f32_16x16x32_f16(false, af, false, bf1,
                                                      (short)0, acc1, false, false);
        __syncthreads();
    }

    // C/D layout: lane holds col n = lane&15, rows m = i + 8*(lane>>4)
    const int mrow  = m0 + wm + ((lane >> 4) << 3);
    const int ncol0 = n0 + wn + (lane & 15);
    const int ncol1 = ncol0 + 16;
    #pragma unroll
    for (int i = 0; i < 8; ++i) {
        int m = mrow + i;
        if (m < Mg) {
            if (ncol0 < Ng)
                atomicAdd(&Cp[(long long)m * c_rs + (long long)ncol0 * c_cs], acc0[i]);
            if (ncol1 < Ng)
                atomicAdd(&Cp[(long long)m * c_rs + (long long)ncol1 * c_cs], acc1[i]);
        }
    }
}

// ---------------- small helper kernels -------------------------------
__global__ void fill0(float* p, long long n) {
    long long i = (long long)blockIdx.x * 256 + threadIdx.x;
    if (i < n) p[i] = 0.f;
}

__global__ void fill2d(float* p, int rows, int cols, int ld) {
    long long i = (long long)blockIdx.x * 256 + threadIdx.x;
    if (i < (long long)rows * cols) {
        int r = (int)(i / cols);
        int c = (int)(i % cols);
        p[(long long)r * ld + c] = 0.f;
    }
}

__global__ void prep_graph(const int* __restrict__ edge, float* deg, float* dinv,
                           float* coef, float* sumnb)
{
    int t = threadIdx.x;
    const int* src = edge;
    const int* dst = edge + EE;
    if (t < PP) { deg[t] = 0.f; sumnb[t] = 0.f; }
    __syncthreads();
    if (t < EE) atomicAdd(&sumnb[src[t]], 1.f);
    if (t < EE + PP) {
        int d = (t < EE) ? dst[t] : (t - EE);
        atomicAdd(&deg[d], 1.f);
    }
    __syncthreads();
    if (t < PP) dinv[t] = rsqrtf(deg[t]);
    __syncthreads();
    if (t < EE + PP) {
        int s = (t < EE) ? src[t] : (t - EE);
        int d = (t < EE) ? dst[t] : (t - EE);
        coef[t] = dinv[s] * dinv[d];
    }
}

__global__ __launch_bounds__(256)
void gcn_scatter(const float* __restrict__ xw, const int* __restrict__ edge,
                 const float* __restrict__ coef, float* __restrict__ out, int C)
{
    int c = blockIdx.x * 256 + threadIdx.x;
    int e = blockIdx.y;
    int b = blockIdx.z;
    if (c >= C) return;
    int s = (e < EE) ? edge[e]      : (e - EE);
    int d = (e < EE) ? edge[EE + e] : (e - EE);
    atomicAdd(&out[((long long)(b * PP + d)) * C + c],
              coef[e] * xw[((long long)(b * PP + s)) * C + c]);
}

__global__ void bias_lrelu_bn(const float* __restrict__ in, const float* __restrict__ bias,
                              const float* __restrict__ s, const float* __restrict__ t,
                              float* __restrict__ out, long long n, int C)
{
    long long i = (long long)blockIdx.x * 256 + threadIdx.x;
    if (i >= n) return;
    int c = (int)(i % C);
    float v = in[i] + bias[c];
    v = (v >= 0.f) ? v : 0.01f * v;
    out[i] = v * s[c] + t[c];
}

// row LayerNorm (+optional pre-bias and post-lrelu); one block per row.
// Row is staged into LDS with gfx1250 async global->LDS DMA when available.
__global__ __launch_bounds__(256)
void layernorm_kernel(const float* __restrict__ in, const float* __restrict__ bias,
                      const float* __restrict__ g, const float* __restrict__ bv,
                      float* __restrict__ out, int C, int lrelu)
{
    __shared__ __align__(16) float row[1024];
    __shared__ float s1[256], s2[256];
    int tid = threadIdx.x;
    const float* x = in + (long long)blockIdx.x * C;

#if USE_ASYNC_LDS
    if (tid * 4 < C) {
        __builtin_amdgcn_global_load_async_to_lds_b128(
            (g_v4i*)(x + tid * 4), (l_v4i*)&row[tid * 4], 0, 0);
    }
    wait_asynccnt0();
    __syncthreads();
#else
    for (int c = tid; c < C; c += 256) row[c] = x[c];
    __syncthreads();
#endif

    float sum = 0.f, sq = 0.f;
    for (int c = tid; c < C; c += 256) {
        float v = row[c] + (bias ? bias[c] : 0.f);
        sum += v; sq += v * v;
    }
    s1[tid] = sum; s2[tid] = sq;
    __syncthreads();
    for (int s = 128; s > 0; s >>= 1) {
        if (tid < s) { s1[tid] += s1[tid + s]; s2[tid] += s2[tid + s]; }
        __syncthreads();
    }
    float mu   = s1[0] / C;
    float var  = s2[0] / C - mu * mu;
    float rstd = rsqrtf(var + 1e-5f);
    for (int c = tid; c < C; c += 256) {
        float v = row[c] + (bias ? bias[c] : 0.f);
        float o = (v - mu) * rstd * g[c] + bv[c];
        if (lrelu) o = (o >= 0.f) ? o : 0.01f * o;
        out[(long long)blockIdx.x * C + c] = o;
    }
}

__global__ __launch_bounds__(256)
void hyp_kernel(const float* __restrict__ h, const float* __restrict__ Wfc,
                const float* __restrict__ bfc, float* __restrict__ hyp)
{
    int b = blockIdx.x;
    __shared__ float sred[256];
    float x = h[b * HH + threadIdx.x];
    for (int j = 0; j < 4; ++j) {
        sred[threadIdx.x] = x * Wfc[threadIdx.x * 4 + j];
        __syncthreads();
        for (int s = 128; s > 0; s >>= 1) {
            if (threadIdx.x < s) sred[threadIdx.x] += sred[threadIdx.x + s];
            __syncthreads();
        }
        if (threadIdx.x == 0) {
            float v = sred[0] + bfc[j];
            v = 1.f / (1.f + expf(-v));
            v = fminf(fmaxf(v, 1e-4f), 0.99f);
            hyp[b * 4 + j] = v;
        }
        __syncthreads();
    }
}

__global__ void y_update(const float* __restrict__ feat, float* __restrict__ y,
                         const float* __restrict__ U, const float* __restrict__ delta,
                         const float* __restrict__ sumnb, const float* __restrict__ hyp,
                         float* __restrict__ outk, float mg, float mv)
{
    int i = blockIdx.x * 256 + threadIdx.x;
    if (i >= BPN) return;
    int n = i & (NN - 1);
    int p = (i >> 10) % PP;
    int b = i / (PP * NN);
    long long r = (long long)(b * PP + p) * TWO_N;
    float alpha = hyp[b * 4 + 0] * 0.01f;
    float tau   = hyp[b * 4 + 1];
    float rho   = hyp[b * 4 + 2];
    float yv = y[i];
    float sg = (yv > 0.f) ? 1.f : ((yv < 0.f) ? -1.f : 0.f);
    float g  = feat[r + n] - feat[r + NN + n] + sg * tau + U[i] * sumnb[p] + delta[i] * rho;
    g = fminf(fmaxf(g, -mg), mg);
    float yn = yv - alpha * g;
    yn = fminf(fmaxf(yn, -mv), mv);
    y[i]    = yn;
    outk[i] = yn;
}

__global__ void delta_scatter(const float* __restrict__ y, const int* __restrict__ edge,
                              float* __restrict__ delta)
{
    int n = blockIdx.x * 256 + threadIdx.x;   // 0..NN-1
    int e = blockIdx.y;                       // 0..EE-1
    int b = blockIdx.z;
    int s = edge[e], d = edge[EE + e];
    float diff = y[(long long)(b * PP + s) * NN + n] - y[(long long)(b * PP + d) * NN + n];
    atomicAdd(&delta[(long long)(b * PP + s) * NN + n],  diff);
    atomicAdd(&delta[(long long)(b * PP + d) * NN + n], -diff);
}

__global__ void u_update(float* __restrict__ U, const float* __restrict__ delta,
                         const float* __restrict__ hyp, float mv)
{
    int i = blockIdx.x * 256 + threadIdx.x;
    if (i >= BPN) return;
    int b = i / (PP * NN);
    float eta = hyp[b * 4 + 3];
    float v = U[i] + delta[i] * eta;
    U[i] = fminf(fmaxf(v, -mv), mv);
}

// ---------------- host orchestration ---------------------------------
static inline dim3 g1(long long n) { return dim3((unsigned)((n + 255) / 256)); }

extern "C" void kernel_launch(void* const* d_in, const int* in_sizes, int n_in,
                              void* d_out, int out_size, void* d_ws, size_t ws_size,
                              hipStream_t stream) {
    (void)in_sizes; (void)n_in; (void)out_size; (void)ws_size;

    const float* A    = (const float*)d_in[0];
    const float* bvec = (const float*)d_in[1];
    const float* Wm[5]  = {(const float*)d_in[2],  (const float*)d_in[6],
                           (const float*)d_in[10], (const float*)d_in[14], (const float*)d_in[18]};
    const float* bm[5]  = {(const float*)d_in[3],  (const float*)d_in[7],
                           (const float*)d_in[11], (const float*)d_in[15], (const float*)d_in[19]};
    const float* bns[5] = {(const float*)d_in[4],  (const float*)d_in[8],
                           (const float*)d_in[12], (const float*)d_in[16], (const float*)d_in[20]};
    const float* bnt[5] = {(const float*)d_in[5],  (const float*)d_in[9],
                           (const float*)d_in[13], (const float*)d_in[17], (const float*)d_in[21]};
    const int Cin[5]  = {TWO_N, HH, 2 * HH, H4, H4};
    const int Cout[5] = {HH, 2 * HH, H4, H4, H4};
    const float* ln_g = (const float*)d_in[22];
    const float* ln_b = (const float*)d_in[23];
    const float* Wd1 = (const float*)d_in[24]; const float* bd1 = (const float*)d_in[25];
    const float* l1g = (const float*)d_in[26]; const float* l1b = (const float*)d_in[27];
    const float* Wd2 = (const float*)d_in[28]; const float* bd2 = (const float*)d_in[29];
    const float* l2g = (const float*)d_in[30]; const float* l2b = (const float*)d_in[31];
    const float* Wd3 = (const float*)d_in[32]; const float* bd3 = (const float*)d_in[33];
    const float* l3g = (const float*)d_in[34]; const float* l3b = (const float*)d_in[35];
    const float* Wfc = (const float*)d_in[36]; const float* bfc = (const float*)d_in[37];
    const int*   edge = (const int*)d_in[38];
    float* out = (float*)d_out;

    float* w = (float*)d_ws;
    size_t off = 0;
    auto alloc = [&](size_t n) { float* p = w + off; off += n; return p; };
    float* AtA  = alloc((size_t)PP * NN * NN);      // 210 MB
    float* feat = alloc((size_t)BB * PP * TWO_N);   // [b*P+p][2N]; cols NN..2N hold Atb
    float* xw   = alloc((size_t)BB * PP * H4);
    float* scat = alloc((size_t)BB * PP * H4);
    float* act0 = alloc((size_t)BB * PP * H4);
    float* act1 = alloc((size_t)BB * PP * H4);
    float* code = alloc((size_t)BB * PP * H4);      // == [B][P*4H]
    float* h1  = alloc(BB * H4);    float* h1n = alloc(BB * H4);
    float* h2  = alloc(BB * 2*HH);  float* h2n = alloc(BB * 2*HH);
    float* h3  = alloc(BB * HH);    float* h3n = alloc(BB * HH);
    float* hyp = alloc(32);
    float* yv  = alloc(BPN);
    float* Uv  = alloc(BPN);
    float* dl  = alloc(BPN);
    float* deg = alloc(64); float* dinv = alloc(64);
    float* coef = alloc(512); float* sumnb = alloc(64);

    auto gemm = [&](const float* Ag, const float* Wg, float* Cg,
                    int Mg, int Ng, int Kg,
                    long long ab, long long ar, long long ac, long long abs_,
                    long long wb, long long wr, long long wc, long long wbs,
                    long long cb, long long cr, long long cc, long long cbs,
                    int batch, int ksplit) {
        dim3 grid((Ng + BN - 1) / BN, (Mg + BM - 1) / BM, batch * ksplit);
        wmma_gemm<<<grid, 256, 0, stream>>>(Ag, Wg, Cg, Mg, Ng, Kg,
                                            ab, ar, ac, abs_, wb, wr, wc, wbs,
                                            cb, cr, cc, cbs, ksplit);
    };

    // -------- per-launch setup --------
    fill0<<<g1((long long)PP * NN * NN), 256, 0, stream>>>(AtA, (long long)PP * NN * NN);
    fill0<<<g1((long long)BB * PP * TWO_N), 256, 0, stream>>>(feat, (long long)BB * PP * TWO_N);
    fill0<<<g1(BPN), 256, 0, stream>>>(yv, BPN);
    fill0<<<g1(BPN), 256, 0, stream>>>(Uv, BPN);
    fill0<<<g1(BPN), 256, 0, stream>>>(dl, BPN);
    prep_graph<<<1, 512, 0, stream>>>(edge, deg, dinv, coef, sumnb);

    // AtA[p] = A0[p]^T @ A0[p] : M=N=1024, K=512, batch=P
    gemm(A, A, AtA, NN, NN, MM,
         0, 1, NN, (long long)MM * NN,
         0, NN, 1, (long long)MM * NN,
         0, NN, 1, (long long)NN * NN,
         PP, 1);

    // Atb -> feat[:, NN:2N] : M=1024(n), N=8(b), K=512(m), batch=P
    gemm(A, bvec, feat, NN, BB, MM,
         0, 1, NN, (long long)MM * NN,
         0, 1, (long long)PP * MM, (long long)MM,
         NN, 1, (long long)PP * TWO_N, (long long)TWO_N,
         PP, 1);

    // -------- K iterations --------
    for (int k = 0; k < KK; ++k) {
        float mg = fmaxf(30.0f, 100.0f - (float)k);
        float mv = fmaxf(10.0f, 200.0f - 3.0f * (float)k);

        // AtAy -> feat[:, 0:NN]
        fill2d<<<g1((long long)BB * PP * NN), 256, 0, stream>>>(feat, BB * PP, NN, TWO_N);
        gemm(AtA, yv, feat, NN, BB, NN,
             0, NN, 1, (long long)NN * NN,
             0, 1, (long long)PP * NN, (long long)NN,
             0, 1, (long long)PP * TWO_N, (long long)TWO_N,
             PP, 1);

        // encoder: 5 GCN conv layers over [B*P, Cin]
        const float* X = feat;
        int x_ld = TWO_N;
        for (int l = 0; l < 5; ++l) {
            long long nxw = (long long)BB * PP * Cout[l];
            fill0<<<g1(nxw), 256, 0, stream>>>(xw, nxw);
            int ks = Cin[l] >= 2048 ? 4 : (Cin[l] >= 1024 ? 2 : 1);
            gemm(X, Wm[l], xw, BB * PP, Cout[l], Cin[l],
                 0, x_ld, 1, 0,
                 0, Cout[l], 1, 0,
                 0, Cout[l], 1, 0,
                 1, ks);
            fill0<<<g1(nxw), 256, 0, stream>>>(scat, nxw);
            gcn_scatter<<<dim3((Cout[l] + 255) / 256, EE + PP, BB), 256, 0, stream>>>(
                xw, edge, coef, scat, Cout[l]);
            float* act = (l & 1) ? act1 : act0;
            bias_lrelu_bn<<<g1(nxw), 256, 0, stream>>>(scat, bm[l], bns[l], bnt[l],
                                                       act, nxw, Cout[l]);
            X = act;
            x_ld = Cout[l];
        }

        // per-(b,p) LayerNorm -> code ([B][P*4H] flat)
        layernorm_kernel<<<BB * PP, 256, 0, stream>>>(X, nullptr, ln_g, ln_b, code, H4, 0);

        // decoder: [8,51200]@Wd1 (split-K 64 over the 210MB weight stream)
        fill0<<<g1(BB * H4), 256, 0, stream>>>(h1, BB * H4);
        gemm(code, Wd1, h1, BB, H4, PP * H4,
             0, (long long)PP * H4, 1, 0,
             0, H4, 1, 0,
             0, H4, 1, 0,
             1, 64);
        layernorm_kernel<<<BB, 256, 0, stream>>>(h1, bd1, l1g, l1b, h1n, H4, 1);

        fill0<<<g1(BB * 2 * HH), 256, 0, stream>>>(h2, BB * 2 * HH);
        gemm(h1n, Wd2, h2, BB, 2 * HH, H4,
             0, H4, 1, 0,
             0, 2 * HH, 1, 0,
             0, 2 * HH, 1, 0,
             1, 8);
        layernorm_kernel<<<BB, 256, 0, stream>>>(h2, bd2, l2g, l2b, h2n, 2 * HH, 1);

        fill0<<<g1(BB * HH), 256, 0, stream>>>(h3, BB * HH);
        gemm(h2n, Wd3, h3, BB, HH, 2 * HH,
             0, 2 * HH, 1, 0,
             0, HH, 1, 0,
             0, HH, 1, 0,
             1, 4);
        layernorm_kernel<<<BB, 256, 0, stream>>>(h3, bd3, l3g, l3b, h3n, HH, 1);

        hyp_kernel<<<BB, 256, 0, stream>>>(h3n, Wfc, bfc, hyp);

        // LASSO state updates + this iteration's output slice
        y_update<<<g1(BPN), 256, 0, stream>>>(feat, yv, Uv, dl, sumnb, hyp,
                                              out + (long long)k * BPN, mg, mv);
        fill0<<<g1(BPN), 256, 0, stream>>>(dl, BPN);
        delta_scatter<<<dim3(NN / 256, EE, BB), 256, 0, stream>>>(yv, edge, dl);
        u_update<<<g1(BPN), 256, 0, stream>>>(Uv, dl, hyp, mv);
    }
}
